// AttentionLayer_48790828482913
// MI455X (gfx1250) — matrix operand order repaired and checked by reading, not compiled, not run on hardware
//
#include <hip/hip_runtime.h>

// Flash-attention forward, B=8, S=2048, D=512, fp32 in/out, gfx1250 (MI455X).
// Compute-bound (AI ~512 FLOP/B) -> bf16 WMMA, f32 accumulate, online softmax.
// 8 waves/WG = 4 M-tiles x 2 D-slices; K/V tiles double-buffered in LDS so
// HBM staging overlaps the WMMA stream; single barrier per key tile.

#define BATCH 8
#define SEQ   2048
#define DIM   512
#define BM    64      // query rows per workgroup (4 M-tiles of 16)
#define BN    32      // keys per inner tile
#define NW    8       // waves per block
#define NKT   (SEQ / BN)

typedef __attribute__((ext_vector_type(8)))  float  v8f;
typedef __attribute__((ext_vector_type(8)))  __bf16 bf8;
typedef __attribute__((ext_vector_type(16))) __bf16 bf16v;

union ABFrag { bf16v v; bf8 h[2]; };

__global__ __launch_bounds__(256, 1)
void fa_fwd_48790828482913(const float* __restrict__ Q,
                           const float* __restrict__ K,
                           const float* __restrict__ V,
                           float* __restrict__ O)
{
    extern __shared__ char smem_raw[];
    __bf16* lQ  = (__bf16*)smem_raw;        // [BM][DIM]          64 KB
    __bf16* lK  = lQ  + BM * DIM;           // [2][BN][DIM]       64 KB (dbl buf)
    __bf16* lVt = lK  + 2 * BN * DIM;       // [2][DIM][BN]       64 KB (dbl buf)
    __bf16* lP  = lVt + 2 * DIM * BN;       // [NW][16][BN]        8 KB

    const int tid   = threadIdx.x;
    const int wave  = tid >> 5;
    const int lane  = tid & 31;
    const int wm    = wave >> 1;            // M tile 0..3
    const int wd    = wave & 1;             // D slice 0..1 (256 cols)
    const int row16 = lane & 15;
    const int hi    = lane >> 4;

    const int nMB   = SEQ / BM;             // 32
    const int batch = blockIdx.x / nMB;
    const int q0    = (blockIdx.x % nMB) * BM;

    const float* Qb = Q + (size_t)batch * SEQ * DIM;
    const float* Kb = K + (size_t)batch * SEQ * DIM;
    const float* Vb = V + (size_t)batch * SEQ * DIM;
    float*       Ob = O + (size_t)batch * SEQ * DIM;

    // ---- stage Q tile (BM x DIM) as bf16 row-major ----
    {
        const float4* src = (const float4*)(Qb + (size_t)q0 * DIM);
        #pragma unroll 4
        for (int it = 0; it < (BM * DIM / 4) / 256; ++it) {   // 32 iters
            const int i = it * 256 + tid;
            float4 f = src[i];
            const int row = i >> 7, c = (i & 127) * 4;
            __bf16* dst = lQ + row * DIM + c;
            dst[0] = (__bf16)f.x; dst[1] = (__bf16)f.y;
            dst[2] = (__bf16)f.z; dst[3] = (__bf16)f.w;
        }
    }

    // ---- stage one K/V tile into LDS buffer `buf` (K row-major, V transposed) ----
    auto stageKV = [&](int kt, int buf) {
        const int k0 = kt * BN;
        const float4* ks = (const float4*)(Kb + (size_t)k0 * DIM);
        const float4* vs = (const float4*)(Vb + (size_t)k0 * DIM);
        __bf16* lKb = lK  + buf * BN * DIM;
        __bf16* lVb = lVt + buf * DIM * BN;
        #pragma unroll 4
        for (int it = 0; it < (BN * DIM / 4) / 256; ++it) {   // 16 iters
            const int i = it * 256 + tid;
            const int row = i >> 7, c = (i & 127) * 4;
            float4 kf = ks[i];
            __bf16* kd = lKb + row * DIM + c;
            kd[0] = (__bf16)kf.x; kd[1] = (__bf16)kf.y;
            kd[2] = (__bf16)kf.z; kd[3] = (__bf16)kf.w;
            float4 vf = vs[i];
            lVb[(c + 0) * BN + row] = (__bf16)vf.x;
            lVb[(c + 1) * BN + row] = (__bf16)vf.y;
            lVb[(c + 2) * BN + row] = (__bf16)vf.z;
            lVb[(c + 3) * BN + row] = (__bf16)vf.w;
        }
    };

    stageKV(0, 0);
    __syncthreads();

    // ---- accumulators: 16 col-tiles x (16x16 f32 C fragment) = 128 VGPRs ----
    const v8f vzero = {0.f, 0.f, 0.f, 0.f, 0.f, 0.f, 0.f, 0.f};
    v8f acc[16];
    #pragma unroll
    for (int j = 0; j < 16; ++j) acc[j] = vzero;

    float rowm[8], rowl[8];
    #pragma unroll
    for (int r = 0; r < 8; ++r) { rowm[r] = -3.0e38f; rowl[r] = 0.f; }

    const __bf16* qrow = lQ + (wm * 16 + row16) * DIM;   // A-frag row base
    __bf16*       pw   = lP + wave * (16 * BN);          // per-wave P bounce

    for (int kt = 0; kt < NKT; ++kt) {
        const int cur = kt & 1;

        // ---- stage NEXT tile into the other buffer (overlaps WMMA stream) ----
        if (kt + 1 < NKT) {
            stageKV(kt + 1, cur ^ 1);
            if (kt + 2 < NKT) {   // pull tile kt+2 toward L2
                __builtin_prefetch(Kb + (size_t)(kt + 2) * BN * DIM + tid * 16, 0, 1);
                __builtin_prefetch(Vb + (size_t)(kt + 2) * BN * DIM + tid * 16, 0, 1);
            }
        }

        const __bf16* lKb   = lK  + cur * BN * DIM;
        const __bf16* lVb   = lVt + cur * DIM * BN;
        const __bf16* kcol0 = lKb + row16 * DIM;          // B-frag keys 0..15
        const __bf16* kcol1 = lKb + (16 + row16) * DIM;   // B-frag keys 16..31

        // ---- S = Q K^T : 16 K-steps (K=32) x 2 N-subtiles ----
        v8f s0 = vzero, s1 = vzero;
        #pragma unroll
        for (int ks = 0; ks < 16; ++ks) {
            ABFrag a, b0, b1;
            const __bf16* ap = qrow + ks * 32 + hi * 8;   // A: lo=K[..+7], hi=+16
            a.h[0] = *(const bf8*)(ap);
            a.h[1] = *(const bf8*)(ap + 16);
            const __bf16* bp0 = kcol0 + ks * 32 + hi * 8;  // B: lo=K[..+7], hi=+16
            b0.h[0] = *(const bf8*)(bp0);
            b0.h[1] = *(const bf8*)(bp0 + 16);
            const __bf16* bp1 = kcol1 + ks * 32 + hi * 8;
            b1.h[0] = *(const bf8*)(bp1);
            b1.h[1] = *(const bf8*)(bp1 + 16);
            s0 = __builtin_amdgcn_wmma_f32_16x16x32_bf16(false, a.v, false, b0.v,
                                                         (short)0, s0, false, false);
            s1 = __builtin_amdgcn_wmma_f32_16x16x32_bf16(false, a.v, false, b1.v,
                                                         (short)0, s1, false, false);
        }

        // ---- online softmax (row = VGPR r + 8*hi, cols striped over 16 lanes) ----
        float sc[8];
        #pragma unroll
        for (int r = 0; r < 8; ++r) {
            float t = fmaxf(s0[r], s1[r]);
            t = fmaxf(t, __shfl_xor(t, 1, 32));
            t = fmaxf(t, __shfl_xor(t, 2, 32));
            t = fmaxf(t, __shfl_xor(t, 4, 32));
            t = fmaxf(t, __shfl_xor(t, 8, 32));
            const float mnew  = fmaxf(rowm[r], t);
            const float scale = __expf(rowm[r] - mnew);
            const float p0 = __expf(s0[r] - mnew);
            const float p1 = __expf(s1[r] - mnew);
            float rs = p0 + p1;
            rs += __shfl_xor(rs, 1, 32);
            rs += __shfl_xor(rs, 2, 32);
            rs += __shfl_xor(rs, 4, 32);
            rs += __shfl_xor(rs, 8, 32);
            rowl[r] = rowl[r] * scale + rs;
            rowm[r] = mnew;
            sc[r]   = scale;
            s0[r] = p0;
            s1[r] = p1;
        }
        #pragma unroll
        for (int j = 0; j < 16; ++j) {
            #pragma unroll
            for (int r = 0; r < 8; ++r) acc[j][r] *= sc[r];
        }

        // ---- P: C-layout -> A-layout bounce through per-wave LDS (bf16) ----
        #pragma unroll
        for (int r = 0; r < 8; ++r) {
            const int rr = r + hi * 8;
            pw[rr * BN + row16]      = (__bf16)s0[r];
            pw[rr * BN + 16 + row16] = (__bf16)s1[r];
        }
        // DS ops are in-order per wave: the loads below observe the stores above.
        ABFrag p;
        const __bf16* pp = pw + row16 * BN + hi * 8;
        p.h[0] = *(const bf8*)(pp);
        p.h[1] = *(const bf8*)(pp + 16);

        // ---- O += P V : one WMMA (K=32 keys) per 16-col output tile ----
        #pragma unroll
        for (int j = 0; j < 16; ++j) {
            ABFrag vb;
            const __bf16* vp = lVb + (wd * 256 + j * 16 + row16) * BN + hi * 8;
            vb.h[0] = *(const bf8*)(vp);
            vb.h[1] = *(const bf8*)(vp + 16);
            acc[j] = __builtin_amdgcn_wmma_f32_16x16x32_bf16(false, p.v, false, vb.v,
                                                             (short)0, acc[j], false, false);
        }

        // one barrier per tile: next iter flips buffers
        __syncthreads();
    }

    // ---- epilogue: normalize by row sum, store fp32 ----
    float invl[8];
    #pragma unroll
    for (int r = 0; r < 8; ++r) invl[r] = 1.0f / rowl[r];

    #pragma unroll
    for (int j = 0; j < 16; ++j) {
        const int col = wd * 256 + j * 16 + row16;
        #pragma unroll
        for (int r = 0; r < 8; ++r) {
            const int rr = r + hi * 8;
            Ob[(size_t)(q0 + wm * 16 + rr) * DIM + col] = acc[j][r] * invl[r];
        }
    }
}

extern "C" void kernel_launch(void* const* d_in, const int* in_sizes, int n_in,
                              void* d_out, int out_size, void* d_ws, size_t ws_size,
                              hipStream_t stream) {
    (void)in_sizes; (void)n_in; (void)out_size; (void)d_ws; (void)ws_size;
    const float* Q = (const float*)d_in[0];
    const float* K = (const float*)d_in[1];
    const float* V = (const float*)d_in[2];
    float*       O = (float*)d_out;

    dim3 grid(BATCH * (SEQ / BM));   // 256 workgroups (~1 per WGP)
    dim3 block(256);                 // 8 waves (wave32), 2 waves/SIMD32
    size_t shmem = (size_t)(BM * DIM + 2 * BN * DIM + 2 * DIM * BN + NW * 16 * BN)
                   * sizeof(__bf16); // 200 KB dynamic LDS (320 KB WGP budget)
    hipLaunchKernelGGL(fa_fwd_48790828482913, grid, block, shmem, stream, Q, K, V, O);
}
